// SigLipLossWithAmbiguityResolution_29635274342861
// MI455X (gfx1250) — matrix operand order repaired and checked by reading, compiled, not run on hardware
//
#include <hip/hip_runtime.h>
#include <stdint.h>

#define S_IMG 16384
#define N_TXT 8192
#define DIM   512

typedef __attribute__((ext_vector_type(16))) __bf16 v16bf;
typedef __attribute__((ext_vector_type(8)))  float  v8f;

#if __has_builtin(__builtin_amdgcn_global_load_async_to_lds_b128) && \
    __has_builtin(__builtin_amdgcn_s_wait_asynccnt)
#define USE_ASYNC_LDS 1
// async builtin operand types: int4 in AS1 (global) / AS3 (LDS)
typedef int v4i __attribute__((vector_size(16)));
typedef __attribute__((address_space(1))) v4i gv4i;
typedef __attribute__((address_space(3))) v4i lv4i;
#else
#define USE_ASYNC_LDS 0
#endif

// round-to-nearest-even float -> bf16 bits
static __device__ __forceinline__ unsigned short f2bf(float x) {
  union { float f; unsigned int u; } c; c.f = x;
  unsigned int r = c.u + 0x7FFFu + ((c.u >> 16) & 1u);
  return (unsigned short)(r >> 16);
}

static __device__ __forceinline__ float softplusf(float x) {
  // numerically stable softplus
  return fmaxf(x, 0.0f) + log1pf(expf(-fabsf(x)));
}

// 256-thread block reduction; returns the sum to all threads
static __device__ __forceinline__ float block_reduce_256(float v, float* sdata, int tid) {
  sdata[tid] = v; __syncthreads();
  #pragma unroll
  for (int off = 128; off > 0; off >>= 1) {
    if (tid < off) sdata[tid] += sdata[tid + off];
    __syncthreads();
  }
  float r = sdata[0]; __syncthreads();
  return r;
}

// ---------------------------------------------------------------- init state
__global__ void init_state(unsigned int* __restrict__ segbits,
                           unsigned int* __restrict__ sel,
                           float* __restrict__ accum) {
  int j = blockIdx.x * blockDim.x + threadIdx.x;
  if (j < N_TXT) { segbits[j] = 0x7F800000u; sel[j] = 0x7FFFFFFFu; }
  if (j == 0) *accum = 0.0f;
}

// ------------------------------------------------- normalize txt -> f32+bf16
__global__ __launch_bounds__(256) void norm_txt(const float* __restrict__ txt,
                                                float* __restrict__ ztxt_out,
                                                unsigned short* __restrict__ ztxt_bf) {
  __shared__ float sdata[256];
  int j = blockIdx.x, tid = threadIdx.x;
  float a = txt[(size_t)j * DIM + tid];
  float b = txt[(size_t)j * DIM + tid + 256];
  float ss = block_reduce_256(a * a + b * b, sdata, tid);
  float scale = 1.0f / (sqrtf(ss) + 1e-12f);
  float z0 = a * scale, z1 = b * scale;
  ztxt_out[(size_t)j * DIM + tid]       = z0;
  ztxt_out[(size_t)j * DIM + tid + 256] = z1;
  ztxt_bf[(size_t)j * DIM + tid]        = f2bf(z0);
  ztxt_bf[(size_t)j * DIM + tid + 256]  = f2bf(z1);
}

// --------------------- normalize img + true-positive logit loss (per image)
__global__ __launch_bounds__(256) void norm_img_tp(const float* __restrict__ img,
                                                   const int* __restrict__ key_idx,
                                                   const float* __restrict__ ztxt,
                                                   const float* __restrict__ tp_ptr,
                                                   const float* __restrict__ b_ptr,
                                                   float* __restrict__ zimg,
                                                   float* __restrict__ loss) {
  __shared__ float sdata[256];
  int i = blockIdx.x, tid = threadIdx.x;
  float a = img[(size_t)i * DIM + tid];
  float c = img[(size_t)i * DIM + tid + 256];
  float ss = block_reduce_256(a * a + c * c, sdata, tid);
  float scale = 1.0f / (sqrtf(ss) + 1e-12f);
  float z0 = a * scale, z1 = c * scale;
  zimg[(size_t)i * DIM + tid]       = z0;
  zimg[(size_t)i * DIM + tid + 256] = z1;
  int k = key_idx[i];
  float t0 = ztxt[(size_t)k * DIM + tid];
  float t1 = ztxt[(size_t)k * DIM + tid + 256];
  float dot = block_reduce_256(z0 * t0 + z1 * t1, sdata, tid);
  if (tid == 0) {
    float t = expf(*tp_ptr);
    float logit = t * dot + *b_ptr;
    loss[i] = softplusf(-logit);
  }
}

// ----------------------------------- segmented min (bit-ordered, loss >= 0)
__global__ void seg_min(const float* __restrict__ loss,
                        const int* __restrict__ key_idx,
                        unsigned int* __restrict__ segbits) {
  int i = blockIdx.x * blockDim.x + threadIdx.x;
  if (i < S_IMG) {
    union { float f; unsigned int u; } c; c.f = loss[i];
    atomicMin(&segbits[key_idx[i]], c.u);
  }
}

// --------------------------- first minimal image index per text key
__global__ void sel_min(const float* __restrict__ loss,
                        const int* __restrict__ key_idx,
                        const unsigned int* __restrict__ segbits,
                        unsigned int* __restrict__ sel) {
  int i = blockIdx.x * blockDim.x + threadIdx.x;
  if (i < S_IMG) {
    union { float f; unsigned int u; } c; c.f = loss[i];
    int k = key_idx[i];
    if (c.u == segbits[k]) atomicMin(&sel[k], (unsigned int)i);
  }
}

// ------------------- gather zimg_selected (f32 out + bf16 for GEMM) + index
__global__ __launch_bounds__(256) void gather_selected(const unsigned int* __restrict__ sel,
                                                       const float* __restrict__ zimg,
                                                       float* __restrict__ zsel_out,
                                                       unsigned short* __restrict__ zsel_bf,
                                                       float* __restrict__ selidx_out) {
  int j = blockIdx.x, tid = threadIdx.x;
  unsigned int sj = sel[j];
  bool valid = sj < (unsigned int)S_IMG;
  const float* src = zimg + (size_t)(valid ? sj : 0u) * DIM;
  float v0 = valid ? src[tid]       : 0.0f;
  float v1 = valid ? src[tid + 256] : 0.0f;
  zsel_out[(size_t)j * DIM + tid]       = v0;
  zsel_out[(size_t)j * DIM + tid + 256] = v1;
  zsel_bf[(size_t)j * DIM + tid]        = f2bf(v0);
  zsel_bf[(size_t)j * DIM + tid + 256]  = f2bf(v1);
  if (tid == 0) selidx_out[j] = valid ? (float)(int)sj : -1.0f;
}

// --------------------------------------------------------------- main GEMM
// C[m,n] = sum_k A[m,k]*B[n,k], A = zimg_selected bf16, B = ztxt bf16.
// 256 threads = 8 waves; block tile 128x128; each wave owns a 32x64 tile:
// 2 A frags x 4 B frags -> 8 accumulators, K-step 32 (v_wmma_f32_16x16x32_bf16).
// Staging: double-buffered LDS fed by GLOBAL_LOAD_ASYNC_TO_LDS_B128 (ASYNCcnt)
// when available; VGPR-staged fallback otherwise.
union Frag16 { v16bf v; uint4 q[2]; };

__global__ __launch_bounds__(256) void gemm_logits_loss(
    const unsigned short* __restrict__ Abf,   // [N_TXT, DIM]
    const unsigned short* __restrict__ Bbf,   // [N_TXT, DIM]
    const float* __restrict__ tp_ptr,
    const float* __restrict__ b_ptr,
    float* __restrict__ logits,               // [N_TXT, N_TXT]
    float* __restrict__ accum) {
  __shared__ unsigned short lds_a[2][128 * 32];
  __shared__ unsigned short lds_b[2][128 * 32];
  __shared__ float sdata[256];

  const int tid  = threadIdx.x;
  const int wave = tid >> 5;
  const int lane = tid & 31;
  const int wm   = wave & 3;     // M strip: 32 rows
  const int wn   = wave >> 2;    // N strip: 64 cols
  const int m0 = blockIdx.y * 128;
  const int n0 = blockIdx.x * 128;

  const float tt = expf(*tp_ptr);
  const float bb = *b_ptr;

  v8f acc[2][4] = {};

  // staging layout: each thread moves 16 halfs (2 x b128) per matrix per K-step
  const int ldr = tid >> 1;          // 0..127 tile row
  const int ldc = (tid & 1) * 16;    // half-offset within 32-K slice
  const unsigned short* gA = Abf + (size_t)(m0 + ldr) * DIM + ldc;
  const unsigned short* gB = Bbf + (size_t)(n0 + ldr) * DIM + ldc;

  auto load_tile = [&](int buf, int kk) {
#if USE_ASYNC_LDS
    __builtin_amdgcn_global_load_async_to_lds_b128(
        (gv4i*)(gA + kk),     (lv4i*)&lds_a[buf][ldr * 32 + ldc],     0, 0);
    __builtin_amdgcn_global_load_async_to_lds_b128(
        (gv4i*)(gA + kk + 8), (lv4i*)&lds_a[buf][ldr * 32 + ldc + 8], 0, 0);
    __builtin_amdgcn_global_load_async_to_lds_b128(
        (gv4i*)(gB + kk),     (lv4i*)&lds_b[buf][ldr * 32 + ldc],     0, 0);
    __builtin_amdgcn_global_load_async_to_lds_b128(
        (gv4i*)(gB + kk + 8), (lv4i*)&lds_b[buf][ldr * 32 + ldc + 8], 0, 0);
#else
    const uint4* pA = (const uint4*)(gA + kk);
    const uint4* pB = (const uint4*)(gB + kk);
    uint4 a0 = pA[0], a1 = pA[1];
    uint4 b0 = pB[0], b1 = pB[1];
    *(uint4*)&lds_a[buf][ldr * 32 + ldc]     = a0;
    *(uint4*)&lds_a[buf][ldr * 32 + ldc + 8] = a1;
    *(uint4*)&lds_b[buf][ldr * 32 + ldc]     = b0;
    *(uint4*)&lds_b[buf][ldr * 32 + ldc + 8] = b1;
#endif
  };
  auto wait_stage = [&]() {
#if USE_ASYNC_LDS
    __builtin_amdgcn_s_wait_asynccnt(0);
#endif
  };

  auto compute = [&](int buf) {
    // A fragments (ISA 16-bit A layout): lane&15 = M, lanes>=16 take K+8/K+24
    Frag16 af[2];
    #pragma unroll
    for (int i = 0; i < 2; ++i) {
      const unsigned short* base =
          &lds_a[buf][(wm * 32 + i * 16 + (lane & 15)) * 32 + (lane >> 4) * 8];
      af[i].q[0] = *(const uint4*)(base);       // K = kb .. kb+7
      af[i].q[1] = *(const uint4*)(base + 16);  // K = kb+16 .. kb+23
    }
    #pragma unroll
    for (int j = 0; j < 4; ++j) {
      // B fragment: lane&15 = N, lanes>=16 take K+8/K+24 (same K order as A)
      Frag16 bf;
      const unsigned short* base =
          &lds_b[buf][(wn * 64 + j * 16 + (lane & 15)) * 32 + (lane >> 4) * 8];
      bf.q[0] = *(const uint4*)(base);
      bf.q[1] = *(const uint4*)(base + 16);
      #pragma unroll
      for (int i = 0; i < 2; ++i) {
        acc[i][j] = __builtin_amdgcn_wmma_f32_16x16x32_bf16(
            /*neg_a=*/false, af[i].v, /*neg_b=*/false, bf.v,
            /*c_mod=*/(short)0, acc[i][j], /*reuse_a=*/false, /*reuse_b=*/false);
      }
    }
  };

  // software pipeline: prefetch next K-slice (async) while computing current
  load_tile(0, 0);
  wait_stage();
  __syncthreads();
  int buf = 0;
  for (int kk = 0; kk < DIM; kk += 32) {
    if (kk + 32 < DIM) load_tile(buf ^ 1, kk + 32);
    compute(buf);
    wait_stage();      // this wave's async fills of buf^1 are in LDS
    __syncthreads();   // all waves done reading buf / done filling buf^1
    buf ^= 1;
  }

  // fused epilogue: scale+bias, store logits, accumulate softplus loss
  float lsum = 0.0f;
  #pragma unroll
  for (int i = 0; i < 2; ++i) {
    const int rowBase = m0 + wm * 32 + i * 16 + ((lane >> 4) * 8);
    #pragma unroll
    for (int j = 0; j < 4; ++j) {
      const int gcol = n0 + wn * 64 + j * 16 + (lane & 15);
      #pragma unroll
      for (int r = 0; r < 8; ++r) {
        int grow = rowBase + r;
        float logit = acc[i][j][r] * tt + bb;
        logits[(size_t)grow * N_TXT + gcol] = logit;
        float z = (grow == gcol) ? logit : -logit;   // label * logit
        lsum += softplusf(-z);
      }
    }
  }
  float bsum = block_reduce_256(lsum, sdata, tid);
  if (tid == 0) atomicAdd(accum, bsum);
}

__global__ void finalize(const float* __restrict__ accum, float* __restrict__ out_loss) {
  *out_loss = *accum / (float)N_TXT;
}

// ---------------------------------------------------------------------------
extern "C" void kernel_launch(void* const* d_in, const int* in_sizes, int n_in,
                              void* d_out, int out_size, void* d_ws, size_t ws_size,
                              hipStream_t stream) {
  (void)in_sizes; (void)n_in; (void)out_size; (void)ws_size;
  const float* img = (const float*)d_in[0];
  const float* txt = (const float*)d_in[1];
  const int*   key = (const int*)d_in[2];
  const float* tp  = (const float*)d_in[3];
  const float* bb  = (const float*)d_in[4];
  float* out = (float*)d_out;

  // output layout: loss(1) | zimg_selected(n*d) | ztxt(n*d) | logits(n*n) | sel_idx(n)
  float* out_loss   = out;
  float* out_zsel   = out + 1;
  float* out_ztxt   = out + 1 + (size_t)N_TXT * DIM;
  float* out_logits = out + 1 + 2 * (size_t)N_TXT * DIM;
  float* out_selidx = out + 1 + 2 * (size_t)N_TXT * DIM + (size_t)N_TXT * N_TXT;

  // workspace layout
  char* ws = (char*)d_ws;
  float*          zimg    = (float*)ws;          ws += (size_t)S_IMG * DIM * 4;
  unsigned short* ztxt_bf = (unsigned short*)ws; ws += (size_t)N_TXT * DIM * 2;
  unsigned short* zsel_bf = (unsigned short*)ws; ws += (size_t)N_TXT * DIM * 2;
  float*          loss    = (float*)ws;          ws += (size_t)S_IMG * 4;
  unsigned int*   segbits = (unsigned int*)ws;   ws += (size_t)N_TXT * 4;
  unsigned int*   sel     = (unsigned int*)ws;   ws += (size_t)N_TXT * 4;
  float*          accum   = (float*)ws;

  init_state<<<(N_TXT + 255) / 256, 256, 0, stream>>>(segbits, sel, accum);
  norm_txt<<<N_TXT, 256, 0, stream>>>(txt, out_ztxt, ztxt_bf);
  norm_img_tp<<<S_IMG, 256, 0, stream>>>(img, key, out_ztxt, tp, bb, zimg, loss);
  seg_min<<<S_IMG / 256, 256, 0, stream>>>(loss, key, segbits);
  sel_min<<<S_IMG / 256, 256, 0, stream>>>(loss, key, segbits, sel);
  gather_selected<<<N_TXT, 256, 0, stream>>>(sel, zimg, out_zsel, zsel_bf, out_selidx);
  gemm_logits_loss<<<dim3(N_TXT / 128, N_TXT / 128), 256, 0, stream>>>(
      zsel_bf, ztxt_bf, tp, bb, out_logits, accum);
  finalize<<<1, 1, 0, stream>>>(accum, out_loss);
}
